// LSTMDQN_23252952940937
// MI455X (gfx1250) — matrix operand order repaired and checked by reading, not compiled, not run on hardware
//
#include <hip/hip_runtime.h>
#include <hip/hip_bf16.h>

// ---------------- problem constants ----------------
#define BB   16          // batch
#define SS   2048        // sequence
#define EMB  256
#define HID  512
#define G3   1536        // 3*HID
#define QHID 256
#define OUTN 8000        // OUT_LEN*ACT_VOCAB

// ---------------- WMMA fragment types ----------------
typedef __bf16  v16bf __attribute__((ext_vector_type(16)));
typedef float   v8f   __attribute__((ext_vector_type(8)));
typedef unsigned int u32x4 __attribute__((ext_vector_type(4)));
typedef int     vsi4  __attribute__((vector_size(16)));   // matches builtin param

union Frag { v16bf v; u32x4 u[2]; };

__device__ __forceinline__ v8f wmma_bf16(v16bf a, v16bf b, v8f c) {
  return __builtin_amdgcn_wmma_f32_16x16x32_bf16(false, a, false, b, (short)0, c,
                                                 false, false);
}

// A fragment (16x32 bf16): lane kh=lid>>4, row=lid&15.
// elems 0..7  -> K = kbase + 8*kh + 0..7
// elems 8..15 -> K = kbase + 16 + 8*kh + 0..7
__device__ __forceinline__ v16bf load_afrag(const __bf16* base, int row, int ld,
                                            int kbase, int kh) {
  Frag f;
  const __bf16* p = base + row * ld + kbase + 8 * kh;
  f.u[0] = *(const u32x4*)p;
  f.u[1] = *(const u32x4*)(p + 16);
  return f.v;
}

// B fragment (32x16 bf16): col=lid&15; elems 0..7 -> K = 8*kh + 0..7,
// elems 8..15 -> K = 16 + 8*kh + 0..7 (p points at K = kbase + 8*kh).
__device__ __forceinline__ v16bf load_bfrag(const __bf16* p, int ld) {
  v16bf v;
#pragma unroll
  for (int s = 0; s < 16; ++s) v[s] = p[(long)(s < 8 ? s : s + 8) * ld];
  return v;
}

__device__ __forceinline__ float sigmoidf_(float x) {
  return 1.0f / (1.0f + __expf(-x));
}

// ---------------- async global->LDS (ASYNCcnt path), guarded ----------------
#if __has_builtin(__builtin_amdgcn_global_load_async_to_lds_b128)
#define HAVE_ASYNC_LDS 1
__device__ __forceinline__ void async_copy16(const float* g, float* l) {
  __builtin_amdgcn_global_load_async_to_lds_b128(
      (__attribute__((address_space(1))) vsi4*)(float*)g,
      (__attribute__((address_space(3))) vsi4*)l, 0, 0);
}
__device__ __forceinline__ void async_wait() {
#if __has_builtin(__builtin_amdgcn_s_wait_asynccnt)
  __builtin_amdgcn_s_wait_asynccnt(0);
#else
  asm volatile("s_wait_asynccnt 0x0" ::: "memory");
#endif
}
#else
#define HAVE_ASYNC_LDS 0
__device__ __forceinline__ void async_copy16(const float* g, float* l) {
  *(u32x4*)l = *(const u32x4*)g;   // synchronous fallback
}
__device__ __forceinline__ void async_wait() {}
#endif

// ---------------- prep: fp32 [N][K] -> bf16 [K][N] ----------------
__global__ void __launch_bounds__(256)
transpose_bf16(const float* __restrict__ in, __bf16* __restrict__ out,
               int N, int K) {
  long total = (long)N * K;
  for (long idx = (long)blockIdx.x * blockDim.x + threadIdx.x; idx < total;
       idx += (long)gridDim.x * blockDim.x) {
    long k = idx / N;
    long n = idx % N;
    out[idx] = (__bf16)in[n * (long)K + k];
  }
}

// ---------------- kernel 1: gi[s][b][g] = emb[tok[b][s]] @ w_ih^T + b_ih ---
// One block per s (M-tile = the 16 batches at timestep s). 8 waves x 12 N-tiles.
__global__ void __launch_bounds__(256)
embed_gemm(const int* __restrict__ tok, const float* __restrict__ emb,
           const __bf16* __restrict__ w_ihT, const float* __restrict__ b_ih,
           float* __restrict__ gi) {
  __shared__ __align__(16) __bf16 A[16 * EMB];   // [b][k] bf16 staging
  const int s   = blockIdx.x;
  const int tid = threadIdx.x;

  // stage one embedding row slice per thread (row = batch, 16 cols each)
  {
    int r  = tid >> 4;
    int c0 = (tid & 15) * 16;
    int t  = tok[r * SS + s];                    // tokens are [B][S][1]
    const float* src = emb + (long)t * EMB + c0;
#pragma unroll
    for (int i = 0; i < 16; ++i) A[r * EMB + c0 + i] = (__bf16)src[i];
  }
  __syncthreads();

  const int lid = tid & 31;
  const int wv  = tid >> 5;
  const int row = lid & 15;
  const int kh  = lid >> 4;
  const int col = lid & 15;

  for (int nt = wv; nt < G3 / 16; nt += 8) {
    v8f acc = {0.f, 0.f, 0.f, 0.f, 0.f, 0.f, 0.f, 0.f};
#pragma unroll
    for (int kc = 0; kc < EMB / 32; ++kc) {
      const int kbase = kc * 32;
      v16bf a = load_afrag(A, row, EMB, kbase, kh);
      v16bf b = load_bfrag(w_ihT + (long)(kbase + 8 * kh) * G3 + nt * 16 + col, G3);
      acc = wmma_bf16(a, b, acc);
    }
    float bias = b_ih[nt * 16 + col];
    float* dst = gi + ((long)s * 16) * G3 + nt * 16 + col;
#pragma unroll
    for (int v = 0; v < 8; ++v)
      dst[(long)(v + 8 * kh) * G3] = acc[v] + bias;
  }
}

// ---------------- kernel 2: persistent GRU scan + q1 head -----------------
// Single 1024-thread workgroup. Wave w owns h columns [16w, 16w+16).
// Dynamic LDS layout:
//   [0      , 32768)  : hf   fp32 h            (16 x 512)
//   [32768  , 49152)  : hb   bf16 h mirror     (16 x 512)
//   [49152  , 147456) : gb0  gi step buffer 0  (16 x 1536 fp32)
//   [147456 , 245760) : gb1  gi step buffer 1
__global__ void __launch_bounds__(1024)
gru_scan(const float* __restrict__ gi, const __bf16* __restrict__ w_hhT,
         const float* __restrict__ b_hh, const __bf16* __restrict__ w_q1T,
         const float* __restrict__ b_q1, __bf16* __restrict__ feat_bf) {
  extern __shared__ __align__(16) char smem[];
  float*  hf  = (float*)smem;
  __bf16* hb  = (__bf16*)(smem + 32768);
  float*  gb0 = (float*)(smem + 49152);
  float*  gb1 = (float*)(smem + 147456);

  const int tid = threadIdx.x;
  const int lid = tid & 31;
  const int w   = tid >> 5;          // wave 0..31 -> j-tile
  const int row = lid & 15;
  const int kh  = lid >> 4;
  const int col = lid & 15;
  const int jb  = w * 16;            // owned column base

  for (int i = tid; i < 16 * HID; i += 1024) {
    hf[i] = 0.0f;
    hb[i] = (__bf16)0.0f;
  }
  // prologue: stage gi[0] into gb0 (6144 x 16B chunks, 6 per thread)
#pragma unroll
  for (int k = 0; k < 6; ++k) {
    int o = (tid + k * 1024) * 4;
    async_copy16(gi + o, gb0 + o);
  }
  async_wait();
  __syncthreads();

  const float br_b = b_hh[jb + col];
  const float bz_b = b_hh[HID + jb + col];
  const float bn_b = b_hh[2 * HID + jb + col];

  for (int t = 0; t < SS; ++t) {
    // kick off async copy of gi[t+1] into the other buffer; it overlaps
    // the entire 48-WMMA K-loop below.
    float* nxt = ((t + 1) & 1) ? gb1 : gb0;
    if (t + 1 < SS) {
      const float* gsrc = gi + (long)(t + 1) * 16 * G3;
#pragma unroll
      for (int k = 0; k < 6; ++k) {
        int o = (tid + k * 1024) * 4;
        async_copy16(gsrc + o, nxt + o);
      }
    }

    v8f accR = {0.f, 0.f, 0.f, 0.f, 0.f, 0.f, 0.f, 0.f};
    v8f accZ = accR, accN = accR;
#pragma unroll 4
    for (int kc = 0; kc < HID / 32; ++kc) {
      const int kbase = kc * 32;
      v16bf a = load_afrag(hb, row, HID, kbase, kh);
      const __bf16* brow = w_hhT + (long)(kbase + 8 * kh) * G3 + col;
      accR = wmma_bf16(a, load_bfrag(brow + jb, G3), accR);
      accZ = wmma_bf16(a, load_bfrag(brow + HID + jb, G3), accZ);
      accN = wmma_bf16(a, load_bfrag(brow + 2 * HID + jb, G3), accN);
    }

    const float* cur = (t & 1) ? gb1 : gb0;   // gi[t], LDS resident
    float nh[8];
#pragma unroll
    for (int v = 0; v < 8; ++v) {
      int m = v + 8 * kh;                     // batch row
      const float* g = cur + m * G3 + jb + col;
      float r  = sigmoidf_(g[0]       + accR[v] + br_b);
      float z  = sigmoidf_(g[HID]     + accZ[v] + bz_b);
      float nn = tanhf    (g[2 * HID] + r * (accN[v] + bn_b));
      float hp = hf[m * HID + jb + col];
      nh[v] = (1.0f - z) * nn + z * hp;
    }
    __syncthreads();                          // all reads of old h done
#pragma unroll
    for (int v = 0; v < 8; ++v) {
      int m = v + 8 * kh;
      hf[m * HID + jb + col] = nh[v];
      hb[m * HID + jb + col] = (__bf16)nh[v];
    }
    async_wait();                             // our gi[t+1] chunks landed
    __syncthreads();                          // new h + full gi[t+1] visible
  }

  // feat = relu(h @ w_q1^T + b_q1) : 16 N-tiles, waves 0..15
  if (w < 16) {
    v8f acc = {0.f, 0.f, 0.f, 0.f, 0.f, 0.f, 0.f, 0.f};
#pragma unroll 4
    for (int kc = 0; kc < HID / 32; ++kc) {
      const int kbase = kc * 32;
      v16bf a = load_afrag(hb, row, HID, kbase, kh);
      v16bf b = load_bfrag(w_q1T + (long)(kbase + 8 * kh) * QHID + w * 16 + col, QHID);
      acc = wmma_bf16(a, b, acc);
    }
    float bias = b_q1[w * 16 + col];
#pragma unroll
    for (int v = 0; v < 8; ++v) {
      int m = v + 8 * kh;
      float f = fmaxf(acc[v] + bias, 0.0f);
      feat_bf[m * QHID + w * 16 + col] = (__bf16)f;
    }
  }
}

// ---------------- kernel 3: q = feat @ w_q2^T + b_q2 ----------------------
__global__ void __launch_bounds__(256)
head_gemm(const __bf16* __restrict__ featb, const __bf16* __restrict__ w_q2T,
          const float* __restrict__ b_q2, float* __restrict__ out) {
  const int gw = (blockIdx.x * 256 + threadIdx.x) >> 5;  // global wave = N-tile
  if (gw >= OUTN / 16) return;                           // wave-uniform exit
  const int lid = threadIdx.x & 31;
  const int row = lid & 15;
  const int kh  = lid >> 4;
  const int col = lid & 15;

  v8f acc = {0.f, 0.f, 0.f, 0.f, 0.f, 0.f, 0.f, 0.f};
#pragma unroll
  for (int kc = 0; kc < QHID / 32; ++kc) {
    const int kbase = kc * 32;
    v16bf a = load_afrag(featb, row, QHID, kbase, kh);
    v16bf b = load_bfrag(w_q2T + (long)(kbase + 8 * kh) * OUTN + gw * 16 + col, OUTN);
    acc = wmma_bf16(a, b, acc);
  }
  float bias = b_q2[gw * 16 + col];
#pragma unroll
  for (int v = 0; v < 8; ++v) {
    int m = v + 8 * kh;
    out[(long)m * OUTN + gw * 16 + col] = acc[v] + bias;
  }
}

// ---------------- host launcher -------------------------------------------
extern "C" void kernel_launch(void* const* d_in, const int* in_sizes, int n_in,
                              void* d_out, int out_size, void* d_ws,
                              size_t ws_size, hipStream_t stream) {
  (void)in_sizes; (void)n_in; (void)out_size; (void)ws_size;
  const int*   tok  = (const int*)d_in[0];
  const float* emb  = (const float*)d_in[1];
  const float* w_ih = (const float*)d_in[2];
  const float* w_hh = (const float*)d_in[3];
  const float* b_ih = (const float*)d_in[4];
  const float* b_hh = (const float*)d_in[5];
  const float* w_q1 = (const float*)d_in[6];
  const float* b_q1 = (const float*)d_in[7];
  const float* w_q2 = (const float*)d_in[8];
  const float* b_q2 = (const float*)d_in[9];
  float* out = (float*)d_out;

  char* ws = (char*)d_ws;
  size_t off = 0;
  auto take = [&](size_t bytes) {
    char* p = ws + off;
    off = (off + bytes + 255) & ~(size_t)255;
    return p;
  };
  __bf16* w_ihT   = (__bf16*)take((size_t)EMB * G3 * 2);
  __bf16* w_hhT   = (__bf16*)take((size_t)HID * G3 * 2);
  __bf16* w_q1T   = (__bf16*)take((size_t)HID * QHID * 2);
  __bf16* w_q2T   = (__bf16*)take((size_t)QHID * OUTN * 2);
  float*  gi      = (float*) take((size_t)SS * 16 * G3 * 4);
  __bf16* feat_bf = (__bf16*)take((size_t)16 * QHID * 2);

  auto blocks = [](long n) { return (int)((n + 255) / 256); };
  transpose_bf16<<<blocks((long)G3 * EMB),   256, 0, stream>>>(w_ih, w_ihT, G3, EMB);
  transpose_bf16<<<blocks((long)G3 * HID),   256, 0, stream>>>(w_hh, w_hhT, G3, HID);
  transpose_bf16<<<blocks((long)QHID * HID), 256, 0, stream>>>(w_q1, w_q1T, QHID, HID);
  transpose_bf16<<<blocks((long)OUTN * QHID),256, 0, stream>>>(w_q2, w_q2T, OUTN, QHID);

  embed_gemm<<<SS, 256, 0, stream>>>(tok, emb, w_ihT, b_ih, gi);
  gru_scan<<<1, 1024, 245760, stream>>>(gi, w_hhT, b_hh, w_q1T, b_q1, feat_bf);
  head_gemm<<<(OUTN / 16 + 7) / 8, 256, 0, stream>>>(feat_bf, w_q2T, b_q2, out);
}
